// SeHGNN_mag_dp_87823491268963
// MI455X (gfx1250) — matrix operand order repaired and checked by reading, not compiled, not run on hardware
//
#include <hip/hip_runtime.h>
#include <hip/hip_bf16.h>
#include <stdint.h>

// ---------------- problem constants ----------------
#define NB    65536      // batch
#define NG    10         // groups / metapaths
#define FD    512        // input feature dim
#define HD    512        // hidden dim
#define NCLS  349        // classes
#define NCP   384        // NCLS padded to 32/BN-tile multiple
#define CD    128        // attention head dim (H/4)
#define GHDIM (NG*HD)    // 5120

// ---------------- WMMA types ----------------
typedef __attribute__((ext_vector_type(16))) __bf16 v16bf;
typedef __attribute__((ext_vector_type(8)))  __bf16 v8bf;
typedef __attribute__((ext_vector_type(8)))  float  v8f;
typedef __attribute__((ext_vector_type(4)))  unsigned int u32x4;
typedef __attribute__((ext_vector_type(8)))  int i32x8;
typedef __attribute__((ext_vector_type(4)))  int i32x4;

// ---------------- GEMM tile config ----------------
// 256 threads = 8 wave32 waves. Block tile 64(M) x 128(N), K-step 32.
// Wave grid 4(M) x 2(N); each wave computes 16x64 via 4 WMMA 16x16 tiles.
#define BM      64
#define BN      128
#define BKK     32
#define APAD    8        // LDS rows 40 elems = 80B: 16B-aligned, conflict-free
#define THREADS 256

// ---------------- staging path selection ----------------
#ifndef USE_TDM
#define USE_TDM 1
#endif
#if defined(__has_builtin)
#  if __has_builtin(__builtin_amdgcn_tensor_load_to_lds)
#    define TDM_AVAILABLE 1
#  endif
#  if __has_builtin(__builtin_amdgcn_s_wait_tensorcnt)
#    define HAVE_WAIT_TENSOR 1
#  endif
#endif
#ifndef TDM_AVAILABLE
#define TDM_AVAILABLE 0
#endif
#ifndef HAVE_WAIT_TENSOR
#define HAVE_WAIT_TENSOR 0
#endif
#define STAGE_TDM (USE_TDM && TDM_AVAILABLE)

// 16-byte global -> LDS async copy (ASYNCcnt pipe) — fallback staging.
__device__ __forceinline__ void copy16_g2l(void* lds, const void* g)
{
    asm volatile("global_load_async_to_lds_b128 %0, %1, off"
                 :: "v"((uint32_t)(uintptr_t)(lds)), "v"(g)
                 : "memory");
}

#if STAGE_TDM
// Tensor Data Mover: DMA a [rows x 32] bf16 2D tile into LDS, padding
// each 16-DWORD row with 4 DWORDs so it lands in our 40-element LDS rows.
__device__ __forceinline__ void tdm_load_tile(void* lds, const void* gsrc,
                                              int64_t rowStrideElems, int rows)
{
    const uint64_t ga = (uint64_t)(uintptr_t)gsrc;
    u32x4 g0;
    g0[0] = 1u;                                   // count=1, user descriptor
    g0[1] = (uint32_t)(uintptr_t)lds;             // lds_addr
    g0[2] = (uint32_t)ga;                         // global_addr[31:0]
    g0[3] = (uint32_t)((ga >> 32) & 0x01FFFFFFu)  // global_addr[56:32]
          | (2u << 30);                           // type=2 ("image")
    i32x8 g1;
    // data_size=2B(code1) | pad_enable | pad_interval=16dw(code3) |
    // pad_amount=4dw(code3)
    g1[0] = (int)((1u << 16) | (1u << 20) | (3u << 22) | (3u << 25));
    const uint32_t dim0 = (uint32_t)rowStrideElems;   // tensor dim0 extent
    const uint32_t dim1 = (uint32_t)rows;             // tensor dim1 extent
    g1[1] = (int)((dim0 & 0xFFFFu) << 16);                        // [63:48]
    g1[2] = (int)((dim0 >> 16) | ((dim1 & 0xFFFFu) << 16));       // [95:64]
    g1[3] = (int)((dim1 >> 16) | ((uint32_t)BKK << 16));          // tile_dim0=32
    g1[4] = (int)((uint32_t)rows);                 // tile_dim1=rows, tile_dim2=0
    const uint64_t st0 = (uint64_t)rowStrideElems; // tensor_dim0_stride
    g1[5] = (int)(uint32_t)st0;
    g1[6] = (int)(uint32_t)((st0 >> 32) & 0xFFFFu);   // stride1 = 0
    g1[7] = 0;
    const i32x4 z4 = {0, 0, 0, 0};                 // groups 2/3 unused (2D)
    const i32x8 z8 = {0, 0, 0, 0, 0, 0, 0, 0};     // extra group (clang-23 form)
    __builtin_amdgcn_tensor_load_to_lds(g0, g1, z4, z4, z8, 0);
}
#endif

__device__ __forceinline__ void wait_stage()
{
#if STAGE_TDM
#  if HAVE_WAIT_TENSOR
    __builtin_amdgcn_s_wait_tensorcnt(0);
#  else
    asm volatile("s_wait_tensorcnt 0x0" ::: "memory");
#  endif
#else
    asm volatile("s_wait_asynccnt 0x0" ::: "memory");
#endif
}

// =====================================================================
// bf16 WMMA GEMM:  C[M,N] = A[M,Kp] * Bt[N,Kp]^T (+bias) (+PReLU)
// A: bf16 row-major, row stride aRow (K-span zero-padded to Kp).
// Bt: bf16 TRANSPOSED weights [Npad][Kp], fully zero-padded -> staging
//     needs no bounds checks. grid.z + group strides give grouped GEMM.
// Double-buffered LDS; staging (TDM or async-b128) overlaps WMMA.
// =====================================================================
template<bool BIAS, bool PRELU, typename OutT>
__global__ __launch_bounds__(THREADS)
void gemm_wmma_bf16(const __bf16* __restrict__ A, int64_t aRow, int64_t aGrp,
                    const __bf16* __restrict__ Bt, int64_t btRow, int64_t bGrp,
                    const float*  __restrict__ bias, int64_t biasGrp,
                    const float*  __restrict__ slope,
                    OutT* __restrict__ C, int64_t cRow, int64_t cGrp,
                    int N, int Kp)
{
    const int g = blockIdx.z;
    A  += (int64_t)g * aGrp;
    Bt += (int64_t)g * bGrp;
    if (BIAS) bias += (int64_t)g * biasGrp;
    C  += (int64_t)g * cGrp;

    const int m0   = blockIdx.y * BM;
    const int n0   = blockIdx.x * BN;
    const int tid  = threadIdx.x;
    const int lane = tid & 31;
    const int wave = tid >> 5;
    const int waveM = (wave & 3) * 16;   // 0,16,32,48
    const int waveN = (wave >> 2) * 64;  // 0,64

    __shared__ __bf16 As[2][BM][BKK + APAD];   // M-major A tiles
    __shared__ __bf16 Bs[2][BN][BKK + APAD];   // N-major (transposed) B tiles

#if !STAGE_TDM
    // 16B-chunk staging coordinates (4 chunks per 32-elem K row)
    const int ar  = tid >> 2;                 // row   0..63
    const int ac  = (tid & 3) * 8;            // K-col 0,8,16,24
    const int br1 = (tid + THREADS) >> 2;     // B rows 64..127
#endif

    v8f acc[4];
#pragma unroll
    for (int s = 0; s < 4; ++s)
#pragma unroll
        for (int e = 0; e < 8; ++e) acc[s][e] = 0.0f;

    const int nsteps = Kp / BKK;

    // ---- prologue: stage tile 0 into buffer 0 ----
#if STAGE_TDM
    if (wave == 0) {
        tdm_load_tile(&As[0][0][0], A  + (int64_t)m0 * aRow,  aRow,  BM);
        tdm_load_tile(&Bs[0][0][0], Bt + (int64_t)n0 * btRow, btRow, BN);
    }
#else
    copy16_g2l(&As[0][ar][ac],  A  + (int64_t)(m0 + ar)  * aRow  + ac);
    copy16_g2l(&Bs[0][ar][ac],  Bt + (int64_t)(n0 + ar)  * btRow + ac);
    copy16_g2l(&Bs[0][br1][ac], Bt + (int64_t)(n0 + br1) * btRow + ac);
#endif

    for (int i = 0; i < nsteps; ++i) {
        wait_stage();
        __syncthreads();

        // ---- issue stage of NEXT tile into the other buffer ----
        if (i + 1 < nsteps) {
            const int nb = (i + 1) & 1;
            const int kn = (i + 1) * BKK;
#if STAGE_TDM
            if (wave == 0) {
                tdm_load_tile(&As[nb][0][0],
                              A  + (int64_t)m0 * aRow  + kn, aRow,  BM);
                tdm_load_tile(&Bs[nb][0][0],
                              Bt + (int64_t)n0 * btRow + kn, btRow, BN);
            }
#else
            copy16_g2l(&As[nb][ar][ac],
                       A  + (int64_t)(m0 + ar)  * aRow  + kn + ac);
            copy16_g2l(&Bs[nb][ar][ac],
                       Bt + (int64_t)(n0 + ar)  * btRow + kn + ac);
            copy16_g2l(&Bs[nb][br1][ac],
                       Bt + (int64_t)(n0 + br1) * btRow + kn + ac);
#endif
        }

        const int buf = i & 1;
        // ---- A fragment 16x32 (ISA 7.12.2): two contiguous 8-elem runs ----
        // lanes 0-15: M=lane,   K = 0..7  and 16..23
        // lanes16-31: M=lane-16,K = 8..15 and 24..31
        const int mrow = waveM + (lane & 15);
        const int akb  = (lane >> 4) ? 8 : 0;
        const v8bf a0  = *(const v8bf*)&As[buf][mrow][akb];
        const v8bf a1  = *(const v8bf*)&As[buf][mrow][akb + 16];
        const v16bf afr = __builtin_shufflevector(
            a0, a1, 0,1,2,3,4,5,6,7,8,9,10,11,12,13,14,15);

        // ---- B fragments 32x16: two 8-elem K runs (same as A) in N-major LDS ----
        const int bkb = (lane >> 4) * 8;
#pragma unroll
        for (int s = 0; s < 4; ++s) {
            const int nrow = waveN + s * 16 + (lane & 15);
            const v8bf b0 = *(const v8bf*)&Bs[buf][nrow][bkb];
            const v8bf b1 = *(const v8bf*)&Bs[buf][nrow][bkb + 16];
            const v16bf bfr = __builtin_shufflevector(
                b0, b1, 0,1,2,3,4,5,6,7,8,9,10,11,12,13,14,15);
            acc[s] = __builtin_amdgcn_wmma_f32_16x16x32_bf16(
                false, afr, false, bfr, (short)0, acc[s], false, false);
        }
        // no trailing barrier: next iteration's wait+barrier protects buffers
    }

    // ---- epilogue: bias / PReLU / store (only N needs a guard) ----
    const float sl = PRELU ? slope[0] : 0.0f;
#pragma unroll
    for (int s = 0; s < 4; ++s) {
        const int col = n0 + waveN + s * 16 + (lane & 15);
        if (col >= N) continue;
        const float bb = BIAS ? bias[col] : 0.0f;
#pragma unroll
        for (int r = 0; r < 8; ++r) {
            // C/D layout: VGPR r -> M=r (lanes 0-15), M=r+8 (lanes 16-31)
            const int row = m0 + waveM + ((lane >> 4) << 3) + r;
            float v = acc[s][r] + bb;
            if (PRELU) v = (v >= 0.0f) ? v : sl * v;
            C[(int64_t)row * cRow + col] = (OutT)v;
        }
    }
}

// =====================================================================
// Weight convert: fp32 [G][K][N] -> bf16 TRANSPOSED padded [G][Np][Kp]
// =====================================================================
__global__ void w_to_bf16_t(const float* __restrict__ in,
                            __bf16* __restrict__ out,
                            int K, int N, int Kp, int Np, int groups)
{
    const int64_t total = (int64_t)groups * Np * Kp;
    int64_t i = (int64_t)blockIdx.x * blockDim.x + threadIdx.x;
    const int64_t stride = (int64_t)gridDim.x * blockDim.x;
    for (; i < total; i += stride) {
        const int64_t g   = i / ((int64_t)Np * Kp);
        const int64_t rem = i % ((int64_t)Np * Kp);
        const int n = (int)(rem / Kp);
        const int k = (int)(rem % Kp);
        __bf16 v = (__bf16)0.0f;
        if (n < N && k < K) v = (__bf16)in[((int64_t)g * K + k) * N + n];
        out[i] = v;
    }
}

// fp32 -> bf16, 8 elements per iteration (n multiple of 8)
__global__ void f32_to_bf16_v(const float* __restrict__ in,
                              __bf16* __restrict__ out, int64_t n8)
{
    int64_t i = (int64_t)blockIdx.x * blockDim.x + threadIdx.x;
    const int64_t stride = (int64_t)gridDim.x * blockDim.x;
    for (; i < n8; i += stride) {
        const float4 f0 = ((const float4*)in)[i * 2];
        const float4 f1 = ((const float4*)in)[i * 2 + 1];
        v8bf o;
        o[0] = (__bf16)f0.x; o[1] = (__bf16)f0.y;
        o[2] = (__bf16)f0.z; o[3] = (__bf16)f0.w;
        o[4] = (__bf16)f1.x; o[5] = (__bf16)f1.y;
        o[6] = (__bf16)f1.z; o[7] = (__bf16)f1.w;
        ((v8bf*)out)[i] = o;
    }
}

// fp32 [R,C] -> bf16 [R,Cp] zero-padded rows
__global__ void f32_to_bf16_pad(const float* __restrict__ in,
                                __bf16* __restrict__ out,
                                int C, int Cp, int64_t R)
{
    const int64_t total = R * Cp;
    int64_t i = (int64_t)blockIdx.x * blockDim.x + threadIdx.x;
    const int64_t stride = (int64_t)gridDim.x * blockDim.x;
    for (; i < total; i += stride) {
        const int64_t r = i / Cp;
        const int     c = (int)(i % Cp);
        out[i] = (c < C) ? (__bf16)in[r * C + c] : (__bf16)0.0f;
    }
}

__global__ void zero_f32(float* __restrict__ p, int n)
{
    int i = blockIdx.x * blockDim.x + threadIdx.x;
    if (i < n) p[i] = 0.0f;
}

// =====================================================================
// Per-sample LayerNorm over [G,H]=5120 jointly + PReLU; bf16 in/out.
// =====================================================================
__global__ __launch_bounds__(THREADS)
void ln_prelu_kernel(const __bf16* __restrict__ in,
                     const float* __restrict__ gain,
                     const float* __restrict__ beta,
                     const float* __restrict__ slope,
                     __bf16* __restrict__ out)
{
    const int b   = blockIdx.x;
    const int tid = threadIdx.x;
    const int NCH = GHDIM / 8; // 640 chunks of 8
    const v8bf* row = (const v8bf*)(in + (int64_t)b * GHDIM);

    float s = 0.0f, ss = 0.0f;
    for (int c = tid; c < NCH; c += THREADS) {
        const v8bf v = row[c];
#pragma unroll
        for (int e = 0; e < 8; ++e) { const float f = (float)v[e]; s += f; ss += f * f; }
    }
#pragma unroll
    for (int off = 16; off > 0; off >>= 1) {
        s  += __shfl_xor(s,  off, 32);
        ss += __shfl_xor(ss, off, 32);
    }
    __shared__ float ps[8], pss[8];
    const int w = tid >> 5, lane = tid & 31;
    if (lane == 0) { ps[w] = s; pss[w] = ss; }
    __syncthreads();
    if (tid == 0) {
        float a = 0.0f, c2 = 0.0f;
#pragma unroll
        for (int i = 0; i < 8; ++i) { a += ps[i]; c2 += pss[i]; }
        const float mean = a / (float)GHDIM;
        const float var  = c2 / (float)GHDIM - mean * mean;
        ps[0]  = mean;
        pss[0] = rsqrtf(var + 1e-5f);
    }
    __syncthreads();
    const float mean = ps[0], inv = pss[0], sl = slope[0];
    v8bf* orow = (v8bf*)(out + (int64_t)b * GHDIM);
    for (int c = tid; c < NCH; c += THREADS) {
        const v8bf v = row[c];
        v8bf o;
#pragma unroll
        for (int e = 0; e < 8; ++e) {
            const int j = c * 8 + e;
            float f = ((float)v[e] - mean) * inv * gain[j] + beta[j];
            o[e] = (__bf16)((f >= 0.0f) ? f : sl * f);
        }
        orow[c] = o;
    }
}

// =====================================================================
// Per-sample attention: scores = relu(q k^T / sqrt(128)); softmax over
// G=10; h3 = gamma * beta @ v + h2. One block per sample.
// =====================================================================
__global__ __launch_bounds__(THREADS)
void attn_kernel(const __bf16* __restrict__ h2,
                 const __bf16* __restrict__ q,
                 const __bf16* __restrict__ k,
                 const __bf16* __restrict__ v,
                 const float* __restrict__ gamma,
                 __bf16* __restrict__ h3)
{
    const int b   = blockIdx.x;
    const int tid = threadIdx.x;
    __shared__ float sq[NG][CD], sk[NG][CD];
    __shared__ float sscore[NG][NG], sbeta[NG][NG];

    const v8bf* qrow = (const v8bf*)(q + (int64_t)b * NG * CD);
    const v8bf* krow = (const v8bf*)(k + (int64_t)b * NG * CD);
    for (int c = tid; c < NG * CD / 8; c += THREADS) {
        const v8bf qv = qrow[c], kv = krow[c];
        const int m = (c * 8) / CD, cc = (c * 8) % CD;
#pragma unroll
        for (int e = 0; e < 8; ++e) {
            sq[m][cc + e] = (float)qv[e];
            sk[m][cc + e] = (float)kv[e];
        }
    }
    __syncthreads();
    if (tid < NG * NG) {
        const int m = tid / NG, n = tid % NG;
        float s = 0.0f;
#pragma unroll 8
        for (int c = 0; c < CD; ++c) s += sq[m][c] * sk[n][c];
        s *= 0.08838834764831845f;            // 1/sqrt(128)
        sscore[m][n] = (s > 0.0f) ? s : 0.0f; // relu on scores
    }
    __syncthreads();
    if (tid < NG) {
        const int m = tid;
        float mx = -1e30f;
        for (int n = 0; n < NG; ++n) mx = fmaxf(mx, sscore[m][n]);
        float e[NG], den = 0.0f;
        for (int n = 0; n < NG; ++n) { e[n] = __expf(sscore[m][n] - mx); den += e[n]; }
        const float r = 1.0f / den;
        for (int n = 0; n < NG; ++n) sbeta[m][n] = e[n] * r;
    }
    __syncthreads();
    const float gm = gamma[0];
    const v8bf* hrow = (const v8bf*)(h2 + (int64_t)b * GHDIM);
    v8bf* orow = (v8bf*)(h3 + (int64_t)b * GHDIM);
    for (int c = tid; c < GHDIM / 8; c += THREADS) {
        const int m = (c * 8) / HD, cc = (c * 8) % HD;
        float o[8];
#pragma unroll
        for (int e = 0; e < 8; ++e) o[e] = 0.0f;
#pragma unroll
        for (int n = 0; n < NG; ++n) {
            const v8bf vv = *(const v8bf*)(v + (int64_t)b * GHDIM + n * HD + cc);
            const float bw = sbeta[m][n];
#pragma unroll
            for (int e = 0; e < 8; ++e) o[e] += bw * (float)vv[e];
        }
        const v8bf hv = hrow[c];
        v8bf res;
#pragma unroll
        for (int e = 0; e < 8; ++e) res[e] = (__bf16)(gm * o[e] + (float)hv[e]);
        orow[c] = res;
    }
}

// =====================================================================
// BatchNorm stats: column-per-thread partials (coalesced) + atomics.
// =====================================================================
__global__ __launch_bounds__(384)
void bn_partial(const float* __restrict__ z, float* __restrict__ sum,
                float* __restrict__ sumsq)
{
    const int j = threadIdx.x;
    if (j >= NCLS) return;
    const int64_t rowsPer = NB / gridDim.x;
    const int64_t r0 = (int64_t)blockIdx.x * rowsPer;
    float s = 0.0f, ss = 0.0f;
    for (int64_t r = r0; r < r0 + rowsPer; ++r) {
        const float v = z[r * NCLS + j];
        s += v; ss += v * v;
    }
    atomicAdd(&sum[j], s);
    atomicAdd(&sumsq[j], ss);
}

__global__ __launch_bounds__(384)
void bn_finalize(const float* __restrict__ sum, const float* __restrict__ sumsq,
                 float* __restrict__ mean, float* __restrict__ inv)
{
    const int j = threadIdx.x;
    if (j >= NCLS) return;
    const float m = sum[j] / (float)NB;
    const float v = sumsq[j] / (float)NB - m * m;
    mean[j] = m;
    inv[j]  = rsqrtf(v + 1e-5f);
}

// =====================================================================
// out = (z - mean) * inv * bn_g + bn_b + lab
// =====================================================================
__global__ void combine_kernel(const float* __restrict__ z,
                               const float* __restrict__ mean,
                               const float* __restrict__ inv,
                               const float* __restrict__ bn_g,
                               const float* __restrict__ bn_b,
                               const float* __restrict__ lab,
                               float* __restrict__ out, int64_t n)
{
    int64_t i = (int64_t)blockIdx.x * blockDim.x + threadIdx.x;
    const int64_t stride = (int64_t)gridDim.x * blockDim.x;
    for (; i < n; i += stride) {
        const int j = (int)(i % NCLS);
        out[i] = (z[i] - mean[j]) * inv[j] * bn_g[j] + bn_b[j] + lab[i];
    }
}

// =====================================================================
// Host-side pipeline
// =====================================================================
extern "C" void kernel_launch(void* const* d_in, const int* in_sizes, int n_in,
                              void* d_out, int out_size, void* d_ws, size_t ws_size,
                              hipStream_t stream)
{
    (void)in_sizes; (void)n_in; (void)out_size; (void)ws_size;
    const float* x         = (const float*)d_in[0];
    /* tgt_feat (d_in[1]) unused: residual=False */
    const float* label_emb = (const float*)d_in[2];
    const float* fp_W1 = (const float*)d_in[3];
    const float* fp_b1 = (const float*)d_in[4];
    const float* ln1_g = (const float*)d_in[5];
    const float* ln1_b = (const float*)d_in[6];
    const float* pr1   = (const float*)d_in[7];
    const float* fp_W2 = (const float*)d_in[8];
    const float* fp_b2 = (const float*)d_in[9];
    const float* ln2_g = (const float*)d_in[10];
    const float* ln2_b = (const float*)d_in[11];
    const float* pr2   = (const float*)d_in[12];
    const float* qW    = (const float*)d_in[13];
    const float* qb    = (const float*)d_in[14];
    const float* kW    = (const float*)d_in[15];
    const float* kb    = (const float*)d_in[16];
    const float* vW    = (const float*)d_in[17];
    const float* vb    = (const float*)d_in[18];
    const float* gamma = (const float*)d_in[19];
    const float* cW    = (const float*)d_in[20];
    const float* cb    = (const float*)d_in[21];
    const float* pr3   = (const float*)d_in[22];
    const float* lrW1  = (const float*)d_in[23];
    const float* lrb1  = (const float*)d_in[24];
    const float* pr4   = (const float*)d_in[25];
    const float* lrW2  = (const float*)d_in[26];
    const float* bn_g  = (const float*)d_in[27];
    const float* bn_b  = (const float*)d_in[28];
    const float* lfW1  = (const float*)d_in[29];
    const float* lfb1  = (const float*)d_in[30];
    const float* pr5   = (const float*)d_in[31];
    const float* lfW2  = (const float*)d_in[32];
    const float* lfb2  = (const float*)d_in[33];

    // ---- workspace carving ----
    char* ws = (char*)d_ws;
    size_t off = 0;
    auto alloc = [&](size_t bytes) -> void* {
        void* p = ws + off;
        off += (bytes + 255) & ~(size_t)255;
        return p;
    };
    const int64_t BGF = (int64_t)NB * NG * FD;
    const int64_t BGH = (int64_t)NB * NG * HD;
    const int64_t BGC = (int64_t)NB * NG * CD;

    // transposed, padded bf16 weights [Npad][Kpad]
    __bf16* W1t  = (__bf16*)alloc((size_t)NG * HD * FD * 2);   // [g][512][512]
    __bf16* W2t  = (__bf16*)alloc((size_t)NG * HD * HD * 2);   // [g][512][512]
    __bf16* qWt  = (__bf16*)alloc((size_t)CD * HD * 2);        // [128][512]
    __bf16* kWt  = (__bf16*)alloc((size_t)CD * HD * 2);        // [128][512]
    __bf16* vWt  = (__bf16*)alloc((size_t)HD * HD * 2);        // [512][512]
    __bf16* cWt  = (__bf16*)alloc((size_t)HD * GHDIM * 2);     // [512][5120]
    __bf16* lr1t = (__bf16*)alloc((size_t)HD * HD * 2);        // [512][512]
    __bf16* lr2t = (__bf16*)alloc((size_t)NCP * HD * 2);       // [384][512]
    __bf16* lf1t = (__bf16*)alloc((size_t)HD * NCP * 2);       // [512][384]
    __bf16* lf2t = (__bf16*)alloc((size_t)NCP * HD * 2);       // [384][512]
    // activations
    __bf16* xb   = (__bf16*)alloc((size_t)BGF * 2);            // x bf16; later h2
    __bf16* t0   = (__bf16*)alloc((size_t)BGH * 2);            // pre-LN h; later h3
    __bf16* t1   = (__bf16*)alloc((size_t)BGH * 2);            // h1; later v
    __bf16* qbuf = (__bf16*)alloc((size_t)BGC * 2);
    __bf16* kbuf = (__bf16*)alloc((size_t)BGC * 2);
    __bf16* z1   = (__bf16*)alloc((size_t)NB * HD * 2);
    __bf16* z2   = (__bf16*)alloc((size_t)NB * HD * 2);
    float*  z3   = (float*) alloc((size_t)NB * NCLS * 4);
    __bf16* leb  = (__bf16*)alloc((size_t)NB * NCP * 2);       // padded [B,384]
    __bf16* lab1 = (__bf16*)alloc((size_t)NB * HD * 2);
    float*  lab2 = (float*) alloc((size_t)NB * NCLS * 4);
    float*  bnsum  = (float*)alloc((size_t)NCLS * 4);
    float*  bnsq   = (float*)alloc((size_t)NCLS * 4);
    float*  bnmean = (float*)alloc((size_t)NCLS * 4);
    float*  bninv  = (float*)alloc((size_t)NCLS * 4);

    const dim3 blk(THREADS);
    const int MR = NB * NG; // 655360 flat rows for q/k/v

    // ---- weight transpose+pad conversions ----
    w_to_bf16_t<<<2048, THREADS, 0, stream>>>(fp_W1, W1t, FD, HD, FD, HD, NG);
    w_to_bf16_t<<<2048, THREADS, 0, stream>>>(fp_W2, W2t, HD, HD, HD, HD, NG);
    w_to_bf16_t<<<256,  THREADS, 0, stream>>>(qW,   qWt,  HD, CD, HD, CD, 1);
    w_to_bf16_t<<<256,  THREADS, 0, stream>>>(kW,   kWt,  HD, CD, HD, CD, 1);
    w_to_bf16_t<<<512,  THREADS, 0, stream>>>(vW,   vWt,  HD, HD, HD, HD, 1);
    w_to_bf16_t<<<2048, THREADS, 0, stream>>>(cW,   cWt,  GHDIM, HD, GHDIM, HD, 1);
    w_to_bf16_t<<<512,  THREADS, 0, stream>>>(lrW1, lr1t, HD, HD, HD, HD, 1);
    w_to_bf16_t<<<512,  THREADS, 0, stream>>>(lrW2, lr2t, HD, NCLS, HD, NCP, 1);
    w_to_bf16_t<<<512,  THREADS, 0, stream>>>(lfW1, lf1t, NCLS, HD, NCP, HD, 1);
    w_to_bf16_t<<<512,  THREADS, 0, stream>>>(lfW2, lf2t, HD, NCLS, HD, NCP, 1);
    // ---- activation conversions ----
    f32_to_bf16_v<<<8192, THREADS, 0, stream>>>(x, xb, BGF / 8);
    f32_to_bf16_pad<<<8192, THREADS, 0, stream>>>(label_emb, leb, NCLS, NCP,
                                                  (int64_t)NB);

    // ---- grouped GEMM 1: h = einsum(bgf,gfh) + fp_b1 -> t0 ----
    {
        dim3 grid(HD / BN, NB / BM, NG);
        gemm_wmma_bf16<true, false, __bf16><<<grid, blk, 0, stream>>>(
            xb, (int64_t)NG * FD, FD,
            W1t, FD, (int64_t)HD * FD,
            fp_b1, HD, nullptr,
            t0, (int64_t)NG * HD, HD, HD, FD);
    }
    ln_prelu_kernel<<<NB, blk, 0, stream>>>(t0, ln1_g, ln1_b, pr1, t1);

    // ---- grouped GEMM 2: t1 -> t0 ----
    {
        dim3 grid(HD / BN, NB / BM, NG);
        gemm_wmma_bf16<true, false, __bf16><<<grid, blk, 0, stream>>>(
            t1, (int64_t)NG * HD, HD,
            W2t, HD, (int64_t)HD * HD,
            fp_b2, HD, nullptr,
            t0, (int64_t)NG * HD, HD, HD, HD);
    }
    ln_prelu_kernel<<<NB, blk, 0, stream>>>(t0, ln2_g, ln2_b, pr2, xb);

    // ---- q / k / v projections over flat [B*G, 512] rows ----
    gemm_wmma_bf16<true, false, __bf16>
        <<<dim3(1, MR / BM, 1), blk, 0, stream>>>(
            xb, HD, 0, qWt, HD, 0, qb, 0, nullptr, qbuf, CD, 0, CD, HD);
    gemm_wmma_bf16<true, false, __bf16>
        <<<dim3(1, MR / BM, 1), blk, 0, stream>>>(
            xb, HD, 0, kWt, HD, 0, kb, 0, nullptr, kbuf, CD, 0, CD, HD);
    gemm_wmma_bf16<true, false, __bf16>
        <<<dim3(HD / BN, MR / BM, 1), blk, 0, stream>>>(
            xb, HD, 0, vWt, HD, 0, vb, 0, nullptr, t1, HD, 0, HD, HD);

    // ---- attention + residual: h3 -> t0 ----
    attn_kernel<<<NB, blk, 0, stream>>>(xb, qbuf, kbuf, t1, gamma, t0);

    // ---- concat projection [B,5120]x[5120,512] + PReLU -> z1 ----
    gemm_wmma_bf16<true, true, __bf16>
        <<<dim3(HD / BN, NB / BM, 1), blk, 0, stream>>>(
            t0, GHDIM, 0, cWt, GHDIM, 0, cb, 0, pr3, z1, HD, 0, HD, GHDIM);

    // ---- lr head: z2 = PReLU(z1*lrW1+b); z3 = z2*lrW2 ----
    gemm_wmma_bf16<true, true, __bf16>
        <<<dim3(HD / BN, NB / BM, 1), blk, 0, stream>>>(
            z1, HD, 0, lr1t, HD, 0, lrb1, 0, pr4, z2, HD, 0, HD, HD);
    gemm_wmma_bf16<false, false, float>
        <<<dim3(NCP / BN, NB / BM, 1), blk, 0, stream>>>(
            z2, HD, 0, lr2t, HD, 0, nullptr, 0, nullptr,
            z3, NCLS, 0, NCLS, HD);

    // ---- BatchNorm statistics (train mode, biased) ----
    zero_f32<<<2, 256, 0, stream>>>(bnsum, NCLS);
    zero_f32<<<2, 256, 0, stream>>>(bnsq, NCLS);
    bn_partial<<<256, 384, 0, stream>>>(z3, bnsum, bnsq);
    bn_finalize<<<1, 384, 0, stream>>>(bnsum, bnsq, bnmean, bninv);

    // ---- label head: lab1 = PReLU(emb*lfW1+b); lab2 = lab1*lfW2+b ----
    gemm_wmma_bf16<true, true, __bf16>
        <<<dim3(HD / BN, NB / BM, 1), blk, 0, stream>>>(
            leb, NCP, 0, lf1t, NCP, 0, lfb1, 0, pr5, lab1, HD, 0, HD, NCP);
    gemm_wmma_bf16<true, false, float>
        <<<dim3(NCP / BN, NB / BM, 1), blk, 0, stream>>>(
            lab1, HD, 0, lf2t, HD, 0, lfb2, 0, nullptr,
            lab2, NCLS, 0, NCLS, HD);

    // ---- final: out = BN(z3) + lab2 ----
    combine_kernel<<<8192, 256, 0, stream>>>(
        z3, bnmean, bninv, bn_g, bn_b, lab2, (float*)d_out,
        (int64_t)NB * NCLS);
}
